// FRKANBioNER_7310034337880
// MI455X (gfx1250) — matrix operand order repaired and checked by reading, not compiled, not run on hardware
//
#include <hip/hip_runtime.h>
#include <hip/hip_bf16.h>

typedef __attribute__((ext_vector_type(16))) _Float16 v16h;
typedef __attribute__((ext_vector_type(8)))  _Float16 v8h;
typedef __attribute__((ext_vector_type(8)))  float    v8f;

#define BB 16
#define LL 512
#define DD 768
#define HH 384
#define H4 1536
#define MM (BB*LL)

// ---------------- utility kernels ----------------
__global__ void fill_f32_kernel(float* p, int n) {
    int i = blockIdx.x * blockDim.x + threadIdx.x;
    if (i < n) p[i] = 0.0f;
}
__global__ void fill_f16_kernel(_Float16* p, int n) {
    int i = blockIdx.x * blockDim.x + threadIdx.x;
    if (i < n) p[i] = (_Float16)0.0f;
}
__global__ void cvt_f32_f16_kernel(const float* __restrict__ s, _Float16* __restrict__ d, int n) {
    int i = blockIdx.x * blockDim.x + threadIdx.x;
    if (i < n) d[i] = (_Float16)s[i];
}
__global__ void bias_sum_kernel(const float* __restrict__ a, const float* __restrict__ b,
                                float* __restrict__ o, int n) {
    int i = blockIdx.x * blockDim.x + threadIdx.x;
    if (i < n) o[i] = a[i] + b[i];
}

// ---------------- valid-token compaction ----------------
// One block per batch row: inclusive scan over L=512 of (valid==1),
// dest = prefix-1 for valid tokens, L (dropped) otherwise.
__global__ void scan_valid_kernel(const int* __restrict__ valid, int* __restrict__ dest) {
    int b = blockIdx.x;
    int t = threadIdx.x;            // 512 threads
    __shared__ int sv[LL];
    int v = (valid[b * LL + t] == 1) ? 1 : 0;
    sv[t] = v;
    __syncthreads();
    for (int off = 1; off < LL; off <<= 1) {
        int x = (t >= off) ? sv[t - off] : 0;
        __syncthreads();
        sv[t] += x;
        __syncthreads();
    }
    dest[b * LL + t] = v ? (sv[t] - 1) : LL;
}

// Scatter valid rows (f32 -> f16) into compacted x16; x16 pre-zeroed.
__global__ void scatter_rows_kernel(const float* __restrict__ src, const int* __restrict__ dest,
                                    _Float16* __restrict__ x16) {
    int m = blockIdx.x;             // b*L + l
    int d = dest[m];
    if (d >= LL) return;
    int b = m >> 9;
    const float* sp = src + (size_t)m * DD;
    _Float16* dp = x16 + (size_t)(b * LL + d) * DD;
    for (int i = threadIdx.x; i < DD; i += blockDim.x) dp[i] = (_Float16)sp[i];
}

// ---------------- WMMA GEMM ----------------
// C[M,N] = A[M,K](f16, row-major) * W[N,K](f16, row-major, i.e. B = W^T) [+ bias[N]]
// One wave (32 threads) computes a 64x64 tile of C as 4x4 WMMA 16x16x32 tiles.
// Fragment layouts follow the CDNA5 ISA 16-bit A/B VGPR tables:
//   A: lane L -> row M = m0+mi*16+(L&15); halves: K = k0+(L>>4)*8 + {0..7, 16..23}
//   B: lane L -> col N = n0+ni*16+(L&15); halves: K = k0+(L>>4)*8 + {0..7, 16..23}
//   C: VGPR g -> row = +g + (L>>4)*8, col = (L&15)
__global__ __launch_bounds__(32)
void gemm_wmma_kernel(const _Float16* __restrict__ A, const _Float16* __restrict__ W,
                      const float* __restrict__ bias, float* __restrict__ C,
                      int M, int N, int K) {
    const int lane = threadIdx.x;
    const int r    = lane & 15;
    const int sel  = lane >> 4;
    const int n0   = blockIdx.x * 64;
    const int m0   = blockIdx.y * 64;

    const _Float16* aBase = A + (size_t)(m0 + r) * K + sel * 8;
    const _Float16* bBase = W + (size_t)(n0 + r) * K + sel * 8;
    const size_t strideA = (size_t)16 * K;

    v8f acc[4][4];
    for (int mi = 0; mi < 4; ++mi)
        for (int ni = 0; ni < 4; ++ni)
            for (int e = 0; e < 8; ++e) acc[mi][ni][e] = 0.0f;

    for (int k0 = 0; k0 < K; k0 += 32) {
        // gfx1250 prefetch of next K-slab (global_prefetch_b8)
        __builtin_prefetch(aBase + k0 + 64, 0, 1);
        __builtin_prefetch(bBase + k0 + 64, 0, 1);

        v16h afr[4], bfr[4];
        for (int mi = 0; mi < 4; ++mi) {
            const _Float16* p = aBase + mi * strideA + k0;
            v8h lo = *(const v8h*)p;          // K = base+0..7
            v8h hi = *(const v8h*)(p + 16);   // K = base+16..23
            for (int e = 0; e < 8; ++e) { afr[mi][e] = lo[e]; afr[mi][8 + e] = hi[e]; }
        }
        for (int ni = 0; ni < 4; ++ni) {
            const _Float16* p = bBase + ni * strideA + k0;
            v8h lo = *(const v8h*)p;          // K = base+0..7
            v8h hi = *(const v8h*)(p + 16);   // K = base+16..23
            for (int e = 0; e < 8; ++e) { bfr[ni][e] = lo[e]; bfr[ni][8 + e] = hi[e]; }
        }
        for (int mi = 0; mi < 4; ++mi)
            for (int ni = 0; ni < 4; ++ni)
                acc[mi][ni] = __builtin_amdgcn_wmma_f32_16x16x32_f16(
                    false, afr[mi], false, bfr[ni], (short)0, acc[mi][ni], false, false);
    }

    for (int mi = 0; mi < 4; ++mi)
        for (int ni = 0; ni < 4; ++ni) {
            const int col = n0 + ni * 16 + r;
            for (int g = 0; g < 8; ++g) {
                const int row = m0 + mi * 16 + sel * 8 + g;
                float v = acc[mi][ni][g];
                if (bias) v += bias[col];
                C[(size_t)row * N + col] = v;
            }
        }
}

// ---------------- LSTM cell update ----------------
// gates[m,:] = h16[m]@Whh^T (from GEMM); add gathered xproj row (x@Wih^T + bias),
// apply i,f,g,o nonlinearities, masked state update.
__global__ void lstm_cell_kernel(const float* __restrict__ gates, const float* __restrict__ xproj,
                                 float* __restrict__ h, float* __restrict__ c,
                                 _Float16* __restrict__ h16, int s, int half, int dir) {
    int idx = blockIdx.x * blockDim.x + threadIdx.x;
    if (idx >= MM * HH) return;
    int m = idx / HH;
    int j = idx - m * HH;
    int l = m & (LL - 1);
    int b = m >> 9;
    int t = (dir == 0) ? (l - half + s) : (l + half - s);
    if (t < 0 || t >= LL) return;   // masked step: (h,c) unchanged
    size_t grow = (size_t)m * H4;
    size_t xrow = (size_t)(b * LL + t) * H4;
    float gi = gates[grow + j]          + xproj[xrow + j];
    float gf = gates[grow + HH + j]     + xproj[xrow + HH + j];
    float gg = gates[grow + 2 * HH + j] + xproj[xrow + 2 * HH + j];
    float go = gates[grow + 3 * HH + j] + xproj[xrow + 3 * HH + j];
    float i_ = 1.0f / (1.0f + __expf(-gi));
    float f_ = 1.0f / (1.0f + __expf(-gf));
    float g_ = tanhf(gg);
    float o_ = 1.0f / (1.0f + __expf(-go));
    float cn = f_ * c[idx] + i_ * g_;
    float hn = o_ * tanhf(cn);
    c[idx] = cn;
    h[idx] = hn;
    h16[idx] = (_Float16)hn;
}

__global__ void copy_h_kernel(const float* __restrict__ h, float* __restrict__ outw, int off) {
    int idx = blockIdx.x * blockDim.x + threadIdx.x;
    if (idx >= MM * HH) return;
    int m = idx / HH;
    int j = idx - m * HH;
    outw[(size_t)m * DD + off + j] = h[idx];
}

// ---------------- attention over 3 window outputs ----------------
__global__ void attn_kernel(const float* __restrict__ outs, float* __restrict__ seq) {
    int m = blockIdx.x;
    int tid = threadIdx.x;          // 256
    const float* o0 = outs + (size_t)m * DD;
    const float* o1 = o0 + (size_t)MM * DD;
    const float* o2 = o1 + (size_t)MM * DD;
    __shared__ float red[3][256];
    float p0 = 0.f, p1 = 0.f, p2 = 0.f;
    for (int i = tid; i < DD; i += 256) {
        float s0 = o0[i], s1 = o1[i], s2 = o2[i];
        float q = s0 + s1 + s2;
        p0 += q * s0; p1 += q * s1; p2 += q * s2;
    }
    red[0][tid] = p0; red[1][tid] = p1; red[2][tid] = p2;
    __syncthreads();
    for (int off = 128; off > 0; off >>= 1) {
        if (tid < off) {
            red[0][tid] += red[0][tid + off];
            red[1][tid] += red[1][tid + off];
            red[2][tid] += red[2][tid + off];
        }
        __syncthreads();
    }
    const float inv = 0.03608439182435161f;  // 1/sqrt(768)
    float d0 = red[0][0] * inv, d1 = red[1][0] * inv, d2 = red[2][0] * inv;
    float mx = fmaxf(d0, fmaxf(d1, d2));
    float e0 = __expf(d0 - mx), e1 = __expf(d1 - mx), e2 = __expf(d2 - mx);
    float rs = 1.0f / (e0 + e1 + e2);
    e0 *= rs; e1 *= rs; e2 *= rs;
    for (int i = tid; i < DD; i += 256) {
        float s0 = o0[i], s1 = o1[i], s2 = o2[i];
        float q = s0 + s1 + s2;
        seq[(size_t)m * DD + i] = q + e0 * s0 + e1 * s1 + e2 * s2;
    }
}

// ---------------- KAN head ----------------
__global__ void kan_kernel(const float* __restrict__ seq, const float* __restrict__ coeffs,
                           const float* __restrict__ kb, float* __restrict__ out) {
    int m = blockIdx.x;
    int tid = threadIdx.x;          // 256
    const float* C0 = coeffs;                   // [11,3,768] cos coeffs
    const float* C1 = coeffs + 11 * 3 * DD;     // [11,3,768] sin coeffs
    float acc[11];
    for (int o = 0; o < 11; ++o) acc[o] = 0.0f;
    for (int i = tid; i < DD; i += 256) {
        float x = seq[(size_t)m * DD + i];
        for (int g = 0; g < 3; ++g) {
            float xg = x * (float)(g + 1);
            float sv, cv;
            __sincosf(xg, &sv, &cv);
            for (int o = 0; o < 11; ++o) {
                int ci = ((o * 3) + g) * DD + i;
                acc[o] += cv * C0[ci] + sv * C1[ci];
            }
        }
    }
    __shared__ float red[256 * 11];
    for (int o = 0; o < 11; ++o) red[tid * 11 + o] = acc[o];
    __syncthreads();
    for (int off = 128; off > 0; off >>= 1) {
        if (tid < off)
            for (int o = 0; o < 11; ++o)
                red[tid * 11 + o] += red[(tid + off) * 11 + o];
        __syncthreads();
    }
    if (tid < 11) out[(size_t)m * 11 + tid] = red[tid] + kb[tid];
}

// ---------------- launch ----------------
extern "C" void kernel_launch(void* const* d_in, const int* in_sizes, int n_in,
                              void* d_out, int out_size, void* d_ws, size_t ws_size,
                              hipStream_t stream) {
    (void)in_sizes; (void)n_in; (void)out_size; (void)ws_size;
    const float* seq_in = (const float*)d_in[0];
    const float* Wih_f  = (const float*)d_in[1];
    const float* Whh_f  = (const float*)d_in[2];
    const float* bih_f  = (const float*)d_in[3];
    const float* bhh_f  = (const float*)d_in[4];
    const float* Wih_b  = (const float*)d_in[5];
    const float* Whh_b  = (const float*)d_in[6];
    const float* bih_b  = (const float*)d_in[7];
    const float* bhh_b  = (const float*)d_in[8];
    const float* kanC   = (const float*)d_in[9];
    const float* kanB   = (const float*)d_in[10];
    const int*   valid  = (const int*)d_in[11];
    float* out = (float*)d_out;

    char* ws = (char*)d_ws;
    size_t woff = 0;
    auto alloc = [&](size_t bytes) -> char* {
        char* p = ws + woff;
        woff += (bytes + 255) & ~(size_t)255;
        return p;
    };
    _Float16* x16   = (_Float16*)alloc((size_t)MM * DD * 2);          // compacted input, f16
    _Float16* wih16 = (_Float16*)alloc((size_t)2 * 3 * H4 * DD * 2);  // [dir][wi][H4][D]
    _Float16* whh16 = (_Float16*)alloc((size_t)2 * 3 * H4 * HH * 2);  // [dir][wi][H4][H]
    float*    bsum  = (float*)  alloc((size_t)2 * 3 * H4 * 4);        // bih+bhh
    int*      dest  = (int*)    alloc((size_t)MM * 4);
    float*    xproj = (float*)  alloc((size_t)MM * H4 * 4);           // x@Wih^T + b
    float*    gates = (float*)  alloc((size_t)MM * H4 * 4);           // h@Whh^T
    float*    hbuf  = (float*)  alloc((size_t)MM * HH * 4);
    float*    cbuf  = (float*)  alloc((size_t)MM * HH * 4);
    _Float16* h16   = (_Float16*)alloc((size_t)MM * HH * 2);
    float*    outs  = (float*)  alloc((size_t)3 * MM * DD * 4);       // per-window bi-LSTM outputs
    float*    seqb  = (float*)  alloc((size_t)MM * DD * 4);

    // 1) compaction: zero x16, prefix-scan valid ids, scatter rows (f32->f16)
    { int n = MM * DD; fill_f16_kernel<<<(n + 255) / 256, 256, 0, stream>>>(x16, n); }
    scan_valid_kernel<<<BB, LL, 0, stream>>>(valid, dest);
    scatter_rows_kernel<<<MM, 256, 0, stream>>>(seq_in, dest, x16);

    // 2) weights -> f16, bias sums
    { int n = 3 * H4 * DD;
      cvt_f32_f16_kernel<<<(n + 255) / 256, 256, 0, stream>>>(Wih_f, wih16, n);
      cvt_f32_f16_kernel<<<(n + 255) / 256, 256, 0, stream>>>(Wih_b, wih16 + n, n); }
    { int n = 3 * H4 * HH;
      cvt_f32_f16_kernel<<<(n + 255) / 256, 256, 0, stream>>>(Whh_f, whh16, n);
      cvt_f32_f16_kernel<<<(n + 255) / 256, 256, 0, stream>>>(Whh_b, whh16 + n, n); }
    { int n = 3 * H4;
      bias_sum_kernel<<<(n + 255) / 256, 256, 0, stream>>>(bih_f, bhh_f, bsum, n);
      bias_sum_kernel<<<(n + 255) / 256, 256, 0, stream>>>(bih_b, bhh_b, bsum + n, n); }

    // 3) windowed bi-LSTMs
    const int wsz[3] = {3, 5, 7};
    dim3 gG(H4 / 64, MM / 64);     // GEMM grid: 24 x 128 waves
    for (int wi = 0; wi < 3; ++wi) {
        int w = wsz[wi], half = w / 2;
        for (int dir = 0; dir < 2; ++dir) {
            const _Float16* Wih16 = wih16 + ((size_t)dir * 3 + wi) * H4 * DD;
            const _Float16* Whh16 = whh16 + ((size_t)dir * 3 + wi) * H4 * HH;
            const float*    bsm   = bsum + ((size_t)dir * 3 + wi) * H4;
            // input projection (shared by all windows at every position)
            gemm_wmma_kernel<<<gG, 32, 0, stream>>>(x16, Wih16, bsm, xproj, MM, H4, DD);
            // zero recurrent state
            { int n = MM * HH;
              fill_f32_kernel<<<(n + 255) / 256, 256, 0, stream>>>(hbuf, n);
              fill_f32_kernel<<<(n + 255) / 256, 256, 0, stream>>>(cbuf, n);
              fill_f16_kernel<<<(n + 255) / 256, 256, 0, stream>>>(h16, n); }
            // recurrence, batched over all 8192 window instances
            for (int s = 0; s < w; ++s) {
                gemm_wmma_kernel<<<gG, 32, 0, stream>>>(h16, Whh16, nullptr, gates, MM, H4, HH);
                int n = MM * HH;
                lstm_cell_kernel<<<(n + 255) / 256, 256, 0, stream>>>(
                    gates, xproj, hbuf, cbuf, h16, s, half, dir);
            }
            int n = MM * HH;
            copy_h_kernel<<<(n + 255) / 256, 256, 0, stream>>>(
                hbuf, outs + (size_t)wi * MM * DD, dir * HH);
        }
    }

    // 4) attention over window outputs, 5) KAN head
    attn_kernel<<<MM, 256, 0, stream>>>(outs, seqb);
    kan_kernel<<<MM, 256, 0, stream>>>(seqb, kanC, kanB, out);
}
